// Swin_59811714564540
// MI455X (gfx1250) — matrix operand order repaired and checked by reading, not compiled, not run on hardware
//
#include <hip/hip_runtime.h>
#include <hip/hip_bf16.h>

typedef __attribute__((ext_vector_type(16))) _Float16 v16h;
typedef __attribute__((ext_vector_type(8)))  _Float16 v8h;
typedef __attribute__((ext_vector_type(4)))  _Float16 v4h;
typedef __attribute__((ext_vector_type(8)))  float    v8f;

// ---------------------------------------------------------------------------
// WMMA helpers (CDNA5 v_wmma_f32_16x16x32_f16, wave32).  All loads are
// UNCONDITIONAL 16B-aligned vector loads; padding rows are handled by
// discarding garbage output rows, never by predicating loads.
// ---------------------------------------------------------------------------
__device__ __forceinline__ v8f wmma32(v16h a, v16h b, v8f c) {
  return __builtin_amdgcn_wmma_f32_16x16x32_f16(false, a, false, b, (short)0, c,
                                                false, false);
}

// A(16x32): lane holds row m = mbase+(lane&15); k = h + 16*(lane>>4)
// -> two contiguous v8h groups at +16*hi and +8.
__device__ __forceinline__ v16h wmma_load_a_lds(const _Float16* buf, int ld,
                                                int mbase, int kbase) {
  const int lane = threadIdx.x & 31;
  const v8h* p = (const v8h*)(buf + (mbase + (lane & 15)) * ld + kbase +
                              16 * (lane >> 4));
  v8h g0 = p[0];
  v8h g1 = p[1];  // +8 halves
  v16h a;
#pragma unroll
  for (int h = 0; h < 8; ++h) { a[h] = g0[h]; a[h + 8] = g1[h]; }
  return a;
}

// B(32x16): lane holds col n = nbase+(lane&15); k = h + 16*(lane>>4).
// Source is n-major: B(k,n) = buf[n*ld + kbase + k] -> 16 contiguous halves.
__device__ __forceinline__ v16h wmma_load_b_lds(const _Float16* buf, int ld,
                                                int nbase, int kbase) {
  const int lane = threadIdx.x & 31;
  const v8h* p = (const v8h*)(buf + (nbase + (lane & 15)) * ld + kbase +
                              16 * (lane >> 4));
  v8h g0 = p[0], g1 = p[1];
  v16h b;
#pragma unroll
  for (int h = 0; h < 8; ++h) { b[h] = g0[h]; b[h + 8] = g1[h]; }
  return b;
}

// B fragment from pre-packed f16 weight matrix W16[rows][ld] (row-major),
// X @ W^T: B(k,n) = W16[rowbase+n][colbase+k] -> 32 contiguous bytes per lane,
// no conversion ALU.
__device__ __forceinline__ v16h wmma_load_b_gmem16(const _Float16* __restrict__ W,
                                                   int ld, int rowbase,
                                                   int colbase) {
  const int lane = threadIdx.x & 31;
  const v8h* p = (const v8h*)(W + (size_t)(rowbase + (lane & 15)) * ld +
                              colbase + 16 * (lane >> 4));
  v8h g0 = p[0], g1 = p[1];
  v16h b;
#pragma unroll
  for (int h = 0; h < 8; ++h) { b[h] = g0[h]; b[h + 8] = g1[h]; }
  return b;
}

// C/D store: lane holds col n; reg r -> row r+8*(lane>>4).  Out-of-range rows
// are redirected to a trash row via address select (no exec manipulation).
__device__ __forceinline__ void wmma_store_c_lds(_Float16* buf, int ld,
                                                 int mbase, int nbase, int mmax,
                                                 v8f c, float mul,
                                                 _Float16* trash) {
  const int lane = threadIdx.x & 31;
  const int hi = lane >> 4;
  const int n = nbase + (lane & 15);
#pragma unroll
  for (int r = 0; r < 8; ++r) {
    int m = mbase + r + 8 * hi;
    _Float16* dst = (m < mmax) ? (buf + m * ld + n) : (trash + lane);
    *dst = (_Float16)(c[r] * mul);
  }
}

__device__ __forceinline__ float hmax16(float v) {
#pragma unroll
  for (int m = 1; m < 16; m <<= 1) v = fmaxf(v, __shfl_xor(v, m, 32));
  return v;
}
__device__ __forceinline__ float hsum16(float v) {
#pragma unroll
  for (int m = 1; m < 16; m <<= 1) v += __shfl_xor(v, m, 32);
  return v;
}
__device__ __forceinline__ float wsum32(float v) {
#pragma unroll
  for (int m = 1; m < 32; m <<= 1) v += __shfl_xor(v, m, 32);
  return v;
}

__device__ __forceinline__ int region_id9(int h, int w) {
  int rh = (h < 49) ? 0 : (h < 53 ? 1 : 2);
  int rw = (w < 49) ? 0 : (w < 53 ? 1 : 2);
  return rh * 3 + rw;
}

// ---------------------------------------------------------------------------
// Kernel 0: pack an f32 weight matrix into f16 (done once per launch; all
// 4096/3136 GEMM workgroups then fetch half the bytes with zero cvt ALU).
// ---------------------------------------------------------------------------
__global__ __launch_bounds__(256) void pack_f16_kernel(
    const float* __restrict__ src, _Float16* __restrict__ dst, int n4) {
  int i = blockIdx.x * 256 + threadIdx.x;
  if (i < n4) {
    float4 f = ((const float4*)src)[i];
    v4h o;
    o[0] = (_Float16)f.x; o[1] = (_Float16)f.y;
    o[2] = (_Float16)f.z; o[3] = (_Float16)f.w;
    ((v4h*)dst)[i] = o;
  }
}

// ---------------------------------------------------------------------------
// Kernel 1: LN1 + shifted-window attention + proj + residual.
// One workgroup (4 waves) per 7x7 window; 4096 windows.
// LDS layout exploits read-overflow: A/B fragment reads of padding rows 49..63
// spill harmlessly into the next buffer (xn->q->k->vT).
// ---------------------------------------------------------------------------
__global__ __launch_bounds__(128) void swin_attn_kernel(
    const float* __restrict__ x, const float* __restrict__ g1,
    const float* __restrict__ b1, const _Float16* __restrict__ qkv_w16,
    const float* __restrict__ qkv_b, const _Float16* __restrict__ proj_w16,
    const float* __restrict__ proj_b, const float* __restrict__ rpb,
    float* __restrict__ y) {
  __shared__ __align__(16) unsigned char smem[65184];
  float*    s_rpb   = (float*)smem;                            // 169*4 f32
  _Float16* s_xn    = (_Float16*)(smem + 2720);                // 49x128 (also attn_out)
  _Float16* s_q     = (_Float16*)(smem + 15264);               // 49x128, pre-scaled
  _Float16* s_k     = (_Float16*)(smem + 27808);               // 49x128
  _Float16* s_vT    = (_Float16*)(smem + 40352);               // 128x64, token tail zeroed
  _Float16* s_p     = (_Float16*)(smem + 56736);               // 64x64 probs
  _Float16* s_trash = (_Float16*)(smem + 64928);               // 128 halves

  const int tid = threadIdx.x, lane = tid & 31, wave = tid >> 5, hi = lane >> 4;
  const int nlo = lane & 15;
  const int wid  = blockIdx.x;
  const int bimg = wid >> 6;
  const int wim  = wid & 63;
  const int wy = wim >> 3, wx = wim & 7;

  for (int i = tid; i < 169 * 4; i += 128) s_rpb[i] = rpb[i];
  {  // zero vT token tail (48..63) so padded-K terms of P@V are exactly 0
    unsigned int* z = (unsigned int*)s_vT;
    for (int i = tid; i < 128 * 8; i += 128) z[(i >> 3) * 32 + 24 + (i & 7)] = 0u;
  }

  // ---- phase 1: gather (roll -SS) + LayerNorm -> f16 LDS -------------------
  for (int t = wave; t < 49; t += 4) {
    int i = t / 7, j = t % 7;
    int hsrc = (wy * 7 + i + 3) % 56;
    int wsrc = (wx * 7 + j + 3) % 56;
    const float* xr = x + ((size_t)bimg * 3136 + hsrc * 56 + wsrc) * 128;
    float4 v = ((const float4*)xr)[lane];
    float s  = v.x + v.y + v.z + v.w;
    float ss = v.x * v.x + v.y * v.y + v.z * v.z + v.w * v.w;
    s = wsum32(s); ss = wsum32(ss);
    float mu   = s * (1.0f / 128.0f);
    float var  = ss * (1.0f / 128.0f) - mu * mu;
    float rstd = rsqrtf(var + 1e-5f);
    int c = lane * 4;
    _Float16* dst = s_xn + t * 128 + c;
    dst[0] = (_Float16)((v.x - mu) * rstd * g1[c + 0] + b1[c + 0]);
    dst[1] = (_Float16)((v.y - mu) * rstd * g1[c + 1] + b1[c + 1]);
    dst[2] = (_Float16)((v.z - mu) * rstd * g1[c + 2] + b1[c + 2]);
    dst[3] = (_Float16)((v.w - mu) * rstd * g1[c + 3] + b1[c + 3]);
  }
  __syncthreads();

  // ---- phase 2: QKV (12 (s3,mt) combos x 8 N-tiles; A loaded once/combo) ---
  const float qscale = 0.17677669529663687f;  // 32^-0.5
  for (int combo = wave; combo < 12; combo += 4) {
    int s3 = combo >> 2, mbase = (combo & 3) * 16;
    v16h a0 = wmma_load_a_lds(s_xn, 128, mbase, 0);
    v16h a1 = wmma_load_a_lds(s_xn, 128, mbase, 32);
    v16h a2 = wmma_load_a_lds(s_xn, 128, mbase, 64);
    v16h a3 = wmma_load_a_lds(s_xn, 128, mbase, 96);
#pragma unroll 1
    for (int nt = 0; nt < 8; ++nt) {
      int nbase = nt * 16;
      float bias = qkv_b[s3 * 128 + nbase + nlo];
      v8f acc;
#pragma unroll
      for (int r = 0; r < 8; ++r) acc[r] = bias;
      acc = wmma32(a0, wmma_load_b_gmem16(qkv_w16, 128, s3 * 128 + nbase, 0),  acc);
      acc = wmma32(a1, wmma_load_b_gmem16(qkv_w16, 128, s3 * 128 + nbase, 32), acc);
      acc = wmma32(a2, wmma_load_b_gmem16(qkv_w16, 128, s3 * 128 + nbase, 64), acc);
      acc = wmma32(a3, wmma_load_b_gmem16(qkv_w16, 128, s3 * 128 + nbase, 96), acc);
      if (s3 == 0) {
        wmma_store_c_lds(s_q, 128, mbase, nbase, 49, acc, qscale, s_trash);
      } else if (s3 == 1) {
        wmma_store_c_lds(s_k, 128, mbase, nbase, 49, acc, 1.0f, s_trash);
      } else {  // V transposed: vT[channel][token]
        int n = nbase + nlo;
#pragma unroll
        for (int r = 0; r < 8; ++r) {
          int m = mbase + r + 8 * hi;
          _Float16* dst = (m < 49) ? (s_vT + n * 64 + m) : (s_trash + lane);
          *dst = (_Float16)acc[r];
        }
      }
    }
  }

  // ---- precompute head-independent bias/mask metadata ----------------------
  const int mt = wave;  // each wave owns rows [16*wave, 16*wave+16)
  int  relx[4][8];
  float madd[4][8];
  bool mok[8], nok[4];
  {
    int im8[8], jm8[8], idm8[8], in4[4], jn4[4], idn4[4];
#pragma unroll
    for (int r = 0; r < 8; ++r) {
      int m = mt * 16 + r + 8 * hi;
      mok[r] = (m < 49);
      int mm = mok[r] ? m : 0;
      im8[r] = mm / 7; jm8[r] = mm % 7;
      idm8[r] = region_id9(wy * 7 + im8[r], wx * 7 + jm8[r]);
    }
#pragma unroll
    for (int nt = 0; nt < 4; ++nt) {
      int n = nt * 16 + nlo;
      nok[nt] = (n < 49);
      int nn = nok[nt] ? n : 0;
      in4[nt] = nn / 7; jn4[nt] = nn % 7;
      idn4[nt] = region_id9(wy * 7 + in4[nt], wx * 7 + jn4[nt]);
    }
#pragma unroll
    for (int nt = 0; nt < 4; ++nt)
#pragma unroll
      for (int r = 0; r < 8; ++r) {
        relx[nt][r] = ((im8[r] - in4[nt] + 6) * 13 + (jm8[r] - jn4[nt] + 6)) * 4;
        madd[nt][r] = (idm8[r] != idn4[nt]) ? -100.0f : 0.0f;
      }
  }
  __syncthreads();

  // ---- phase 3: per-head S + bias + mask + softmax + P@V -------------------
  for (int head = 0; head < 4; ++head) {
    float sc[4][8];
    {
      v16h a = wmma_load_a_lds(s_q, 128, mt * 16, head * 32);
#pragma unroll
      for (int nt = 0; nt < 4; ++nt) {
        v16h b = wmma_load_b_lds(s_k, 128, nt * 16, head * 32);
        v8f z;
#pragma unroll
        for (int r = 0; r < 8; ++r) z[r] = 0.0f;
        v8f c = wmma32(a, b, z);
#pragma unroll
        for (int r = 0; r < 8; ++r) sc[nt][r] = c[r];
      }
    }
#pragma unroll
    for (int nt = 0; nt < 4; ++nt)
#pragma unroll
      for (int r = 0; r < 8; ++r) {
        float t = sc[nt][r] + s_rpb[relx[nt][r] + head] + madd[nt][r];
        sc[nt][r] = (mok[r] && nok[nt]) ? t : -1e30f;
      }
#pragma unroll
    for (int r = 0; r < 8; ++r) {  // register softmax over 64 cols
      float mx = fmaxf(fmaxf(sc[0][r], sc[1][r]), fmaxf(sc[2][r], sc[3][r]));
      mx = hmax16(mx);
      float sm = 0.0f;
#pragma unroll
      for (int nt = 0; nt < 4; ++nt) {
        sc[nt][r] = __expf(sc[nt][r] - mx);
        sm += sc[nt][r];
      }
      sm = hsum16(sm);
      float inv = 1.0f / sm;
#pragma unroll
      for (int nt = 0; nt < 4; ++nt) sc[nt][r] *= inv;
    }
#pragma unroll
    for (int nt = 0; nt < 4; ++nt)
#pragma unroll
      for (int r = 0; r < 8; ++r)
        s_p[(mt * 16 + r + 8 * hi) * 64 + nt * 16 + nlo] = (_Float16)sc[nt][r];
    __syncthreads();
    {
      v16h pa0 = wmma_load_a_lds(s_p, 64, mt * 16, 0);
      v16h pa1 = wmma_load_a_lds(s_p, 64, mt * 16, 32);
#pragma unroll
      for (int nt2 = 0; nt2 < 2; ++nt2) {
        v8f acc;
#pragma unroll
        for (int r = 0; r < 8; ++r) acc[r] = 0.0f;
        acc = wmma32(pa0, wmma_load_b_lds(s_vT, 64, head * 32 + nt2 * 16, 0),  acc);
        acc = wmma32(pa1, wmma_load_b_lds(s_vT, 64, head * 32 + nt2 * 16, 32), acc);
        wmma_store_c_lds(s_xn /*attn_out*/, 128, mt * 16, head * 32 + nt2 * 16,
                         49, acc, 1.0f, s_trash);
      }
    }
    __syncthreads();
  }

  // ---- phase 4: proj + residual, scatter through inverse roll --------------
  {
    v16h a0 = wmma_load_a_lds(s_xn, 128, mt * 16, 0);
    v16h a1 = wmma_load_a_lds(s_xn, 128, mt * 16, 32);
    v16h a2 = wmma_load_a_lds(s_xn, 128, mt * 16, 64);
    v16h a3 = wmma_load_a_lds(s_xn, 128, mt * 16, 96);
    size_t gbase[8];
#pragma unroll
    for (int r = 0; r < 8; ++r) {
      int m = mt * 16 + r + 8 * hi;
      int mm = (m < 49) ? m : 0;
      int i = mm / 7, j = mm % 7;
      int hsrc = (wy * 7 + i + 3) % 56;
      int wsrc = (wx * 7 + j + 3) % 56;
      gbase[r] = ((size_t)bimg * 3136 + hsrc * 56 + wsrc) * 128;
    }
#pragma unroll 1
    for (int nt = 0; nt < 8; ++nt) {
      float pb = proj_b[nt * 16 + nlo];
      v8f acc;
#pragma unroll
      for (int r = 0; r < 8; ++r) acc[r] = pb;
      acc = wmma32(a0, wmma_load_b_gmem16(proj_w16, 128, nt * 16, 0),  acc);
      acc = wmma32(a1, wmma_load_b_gmem16(proj_w16, 128, nt * 16, 32), acc);
      acc = wmma32(a2, wmma_load_b_gmem16(proj_w16, 128, nt * 16, 64), acc);
      acc = wmma32(a3, wmma_load_b_gmem16(proj_w16, 128, nt * 16, 96), acc);
#pragma unroll
      for (int r = 0; r < 8; ++r) {
        if (mok[r]) {
          size_t gi = gbase[r] + nt * 16 + nlo;
          y[gi] = x[gi] + acc[r];
        }
      }
    }
  }
}

// ---------------------------------------------------------------------------
// Kernel 2: LN2 + MLP (128->512 GELU ->128) + residual, in-place on y.
// One workgroup per 64 tokens; hidden dim chunked by 128 (stays in LDS).
// ---------------------------------------------------------------------------
__global__ __launch_bounds__(128) void swin_mlp_kernel(
    float* __restrict__ y, const float* __restrict__ g2,
    const float* __restrict__ b2, const _Float16* __restrict__ fc1_w16,
    const float* __restrict__ fc1_b, const _Float16* __restrict__ fc2_w16,
    const float* __restrict__ fc2_b) {
  __shared__ __align__(16) _Float16 s_xn[64 * 128];
  __shared__ __align__(16) _Float16 s_h[64 * 128];
  const int tid = threadIdx.x, lane = tid & 31, wave = tid >> 5, hi = lane >> 4;
  const int nlo = lane & 15;
  const size_t row0 = (size_t)blockIdx.x * 64;

  for (int t = wave; t < 64; t += 4) {
    const float* xr = y + (row0 + t) * 128;
    float4 v = ((const float4*)xr)[lane];
    float s  = v.x + v.y + v.z + v.w;
    float ss = v.x * v.x + v.y * v.y + v.z * v.z + v.w * v.w;
    s = wsum32(s); ss = wsum32(ss);
    float mu   = s * (1.0f / 128.0f);
    float var  = ss * (1.0f / 128.0f) - mu * mu;
    float rstd = rsqrtf(var + 1e-5f);
    int c = lane * 4;
    _Float16* dst = s_xn + t * 128 + c;
    dst[0] = (_Float16)((v.x - mu) * rstd * g2[c + 0] + b2[c + 0]);
    dst[1] = (_Float16)((v.y - mu) * rstd * g2[c + 1] + b2[c + 1]);
    dst[2] = (_Float16)((v.z - mu) * rstd * g2[c + 2] + b2[c + 2]);
    dst[3] = (_Float16)((v.w - mu) * rstd * g2[c + 3] + b2[c + 3]);
  }
  __syncthreads();

  const int mt = wave;
  // xn2 A-fragments are invariant across the whole kernel: load once.
  v16h xa0 = wmma_load_a_lds(s_xn, 128, mt * 16, 0);
  v16h xa1 = wmma_load_a_lds(s_xn, 128, mt * 16, 32);
  v16h xa2 = wmma_load_a_lds(s_xn, 128, mt * 16, 64);
  v16h xa3 = wmma_load_a_lds(s_xn, 128, mt * 16, 96);

  float acc[8][8];
#pragma unroll
  for (int nt = 0; nt < 8; ++nt)
#pragma unroll
    for (int r = 0; r < 8; ++r) acc[nt][r] = 0.0f;

  for (int hc = 0; hc < 4; ++hc) {
    // h_chunk = GELU(xn2 @ fc1_w[hc*128..]^T + b)
#pragma unroll 1
    for (int nt = 0; nt < 8; ++nt) {
      float fb = fc1_b[hc * 128 + nt * 16 + nlo];
      v8f h;
#pragma unroll
      for (int r = 0; r < 8; ++r) h[r] = fb;
      h = wmma32(xa0, wmma_load_b_gmem16(fc1_w16, 128, hc * 128 + nt * 16, 0),  h);
      h = wmma32(xa1, wmma_load_b_gmem16(fc1_w16, 128, hc * 128 + nt * 16, 32), h);
      h = wmma32(xa2, wmma_load_b_gmem16(fc1_w16, 128, hc * 128 + nt * 16, 64), h);
      h = wmma32(xa3, wmma_load_b_gmem16(fc1_w16, 128, hc * 128 + nt * 16, 96), h);
#pragma unroll
      for (int r = 0; r < 8; ++r) {
        float v = h[r];
        v = 0.5f * v * (1.0f + erff(v * 0.70710678118f));  // exact GELU
        s_h[(mt * 16 + r + 8 * hi) * 128 + nt * 16 + nlo] = (_Float16)v;
      }
    }
    __syncthreads();
    // acc += h_chunk @ fc2_w[:, hc*128..]^T   (A fragments loaded once/chunk)
    v16h ha0 = wmma_load_a_lds(s_h, 128, mt * 16, 0);
    v16h ha1 = wmma_load_a_lds(s_h, 128, mt * 16, 32);
    v16h ha2 = wmma_load_a_lds(s_h, 128, mt * 16, 64);
    v16h ha3 = wmma_load_a_lds(s_h, 128, mt * 16, 96);
#pragma unroll 1
    for (int nt = 0; nt < 8; ++nt) {
      v8f c;
#pragma unroll
      for (int r = 0; r < 8; ++r) c[r] = acc[nt][r];
      c = wmma32(ha0, wmma_load_b_gmem16(fc2_w16, 512, nt * 16, hc * 128 + 0),  c);
      c = wmma32(ha1, wmma_load_b_gmem16(fc2_w16, 512, nt * 16, hc * 128 + 32), c);
      c = wmma32(ha2, wmma_load_b_gmem16(fc2_w16, 512, nt * 16, hc * 128 + 64), c);
      c = wmma32(ha3, wmma_load_b_gmem16(fc2_w16, 512, nt * 16, hc * 128 + 96), c);
#pragma unroll
      for (int r = 0; r < 8; ++r) acc[nt][r] = c[r];
    }
    __syncthreads();
  }

#pragma unroll
  for (int nt = 0; nt < 8; ++nt)
#pragma unroll
    for (int r = 0; r < 8; ++r) {
      size_t gi = (row0 + mt * 16 + r + 8 * hi) * 128 + nt * 16 + nlo;
      y[gi] = acc[nt][r] + fc2_b[nt * 16 + nlo] + y[gi];
    }
}

// ---------------------------------------------------------------------------
extern "C" void kernel_launch(void* const* d_in, const int* in_sizes, int n_in,
                              void* d_out, int out_size, void* d_ws,
                              size_t ws_size, hipStream_t stream) {
  (void)in_sizes; (void)n_in; (void)ws_size; (void)out_size;
  const float* x      = (const float*)d_in[0];
  const float* n1g    = (const float*)d_in[1];
  const float* n1b    = (const float*)d_in[2];
  const float* qkv_w  = (const float*)d_in[3];
  const float* qkv_b  = (const float*)d_in[4];
  const float* proj_w = (const float*)d_in[5];
  const float* proj_b = (const float*)d_in[6];
  const float* rpb    = (const float*)d_in[7];
  const float* n2g    = (const float*)d_in[8];
  const float* n2b    = (const float*)d_in[9];
  const float* fc1_w  = (const float*)d_in[10];
  const float* fc1_b  = (const float*)d_in[11];
  const float* fc2_w  = (const float*)d_in[12];
  const float* fc2_b  = (const float*)d_in[13];
  float* out = (float*)d_out;

  // f16 weight cache in workspace (re-packed every call; deterministic).
  _Float16* qkv_w16  = (_Float16*)d_ws;                  // 384*128
  _Float16* proj_w16 = qkv_w16 + 384 * 128;              // 128*128
  _Float16* fc1_w16  = proj_w16 + 128 * 128;             // 512*128
  _Float16* fc2_w16  = fc1_w16 + 512 * 128;              // 128*512
  pack_f16_kernel<<<(384 * 128 / 4 + 255) / 256, 256, 0, stream>>>(qkv_w, qkv_w16, 384 * 128 / 4);
  pack_f16_kernel<<<(128 * 128 / 4 + 255) / 256, 256, 0, stream>>>(proj_w, proj_w16, 128 * 128 / 4);
  pack_f16_kernel<<<(512 * 128 / 4 + 255) / 256, 256, 0, stream>>>(fc1_w, fc1_w16, 512 * 128 / 4);
  pack_f16_kernel<<<(128 * 512 / 4 + 255) / 256, 256, 0, stream>>>(fc2_w, fc2_w16, 128 * 512 / 4);

  swin_attn_kernel<<<4096, 128, 0, stream>>>(x, n1g, n1b, qkv_w16, qkv_b,
                                             proj_w16, proj_b, rpb, out);
  swin_mlp_kernel<<<3136, 128, 0, stream>>>(out, n2g, n2b, fc1_w16, fc1_b,
                                            fc2_w16, fc2_b);
}
